// Quant3Linear_3796751090063
// MI455X (gfx1250) — matrix operand order repaired and checked by reading, not compiled, not run on hardware
//
#include <hip/hip_runtime.h>
#include <hip/hip_bf16.h>
#include <stdint.h>

#define INF   8192
#define OUTF  8192
#define NOUT  64
#define COLS_PER_BLOCK 128   // 8 waves * 16 columns

typedef _Float16 v16h __attribute__((ext_vector_type(16)));
typedef _Float16 v8h  __attribute__((ext_vector_type(8)));
typedef float    v8f  __attribute__((ext_vector_type(8)));

// ---------------------------------------------------------------------------
// Kernel 1: partial quantized dot  part[kb][col] = sum_{i in slice} x[i]*q[i,col]
// Wave layout: each wave owns 16 output columns. lane = h*16 + n,
//   n = column within tile, h = K-half (0: K 0..15 of a 32-group, 1: K 16..31).
// 3-bit weight k of a group lives at bits [3k,3k+3) of concat(r0,r1,r2), so a
// lane loads rows (3g+h, 3g+h+1), makes a 64-bit word, shifts by 16h once and
// then every extract is a constant-shift bfe.
// B(f16) built exactly via as_f16(0x6400|q) = 1024+q, minus 1024 (packed).
// A(f16) = x slice replicated over all 16 rows -> every row of D equals the
// 16 column dots; lanes 0..15 read D row0 (c[0]).
// ---------------------------------------------------------------------------
__global__ __launch_bounds__(256) void quant3_partial_kernel(
    const uint32_t* __restrict__ qw,   // [3*INF/32, OUTF]
    const float*    __restrict__ x,    // [INF]
    float*          __restrict__ part, // [kpart, OUTF]
    int kper)                          // inputs per K-slice (multiple of 32)
{
    __shared__ __align__(16) _Float16 xh[INF]; // f16 copy of this block's x slice

    const int tid  = threadIdx.x;
    const int lane = tid & 31;
    const int h    = lane >> 4;      // K-half
    const int n    = lane & 15;      // column within wave tile
    const int wave = tid >> 5;
    const int col  = blockIdx.x * COLS_PER_BLOCK + wave * 16 + n;
    const int kb   = blockIdx.y;
    const int xbase = kb * kper;

    // stage x slice as f16 into LDS
    for (int i = tid; i < kper; i += 256)
        xh[i] = (_Float16)x[xbase + i];
    __syncthreads();

    const int ngroups = kper >> 5;           // 32 inputs per group
    const int g0 = kb * ngroups;

    v8f acc = {};
    const uint32_t* qp = qw + (size_t)(3 * g0 + h) * OUTF + col;

    for (int t = 0; t < ngroups; ++t) {
        uint32_t ra = qp[0];
        uint32_t rb = qp[OUTF];
        __builtin_prefetch(qp + 3 * OUTF, 0, 1);   // next group's rows
        qp += 3 * OUTF;

        uint64_t v = ((uint64_t)rb << 32) | (uint64_t)ra;
        v = h ? (v >> 16) : v;                     // branch-free select

        // B matrix: 16 exact f16 weights for this column / K-half
        v16h b;
#pragma unroll
        for (int e = 0; e < 16; ++e) {
            uint32_t q3 = (uint32_t)(v >> (3 * e)) & 7u;
            unsigned short bits = (unsigned short)(0x6400u | q3); // 1024+q exactly
            b[e] = __builtin_bit_cast(_Float16, bits);
        }
        b = b - (_Float16)1024.0f;                 // exact, packed f16 subs

        // A matrix: x replicated across rows. Element order, same as B:
        // el 0..15 = K 16h..16h+15.
        const _Float16* xp = xh + (size_t)t * 32 + 16 * h;
        v8h alo = *(const v8h*)(xp);
        v8h ahi = *(const v8h*)(xp + 8);
        v16h a = __builtin_shufflevector(alo, ahi,
                    0,1,2,3,4,5,6,7,8,9,10,11,12,13,14,15);

        acc = __builtin_amdgcn_wmma_f32_16x16x32_f16(
                  false, a, false, b, (short)0, acc, false, false);
    }

    if (h == 0)
        part[(size_t)kb * OUTF + col] = acc[0];   // D row 0, column n
}

// ---------------------------------------------------------------------------
// Kernel 2: finalize. Deterministic tree reduction for S = sum(x), combine
// K-partials, apply scale/zero/bias, add outlier GEMV via LDS-staged x_out.
// ---------------------------------------------------------------------------
__global__ __launch_bounds__(256) void quant3_finalize_kernel(
    const float* __restrict__ x,
    const float* __restrict__ scales,
    const float* __restrict__ zeros,
    const float* __restrict__ bias,
    const float* __restrict__ ow,     // [OUTF, NOUT]
    const int*   __restrict__ oidx,   // [NOUT]
    const float* __restrict__ part,   // [kpart, OUTF]
    int kpart,
    float*       __restrict__ out)
{
    __shared__ float sred[256];
    __shared__ float xo[NOUT];

    const int tid = threadIdx.x;

    float s = 0.f;
    for (int i = tid; i < INF; i += 256) s += x[i];
    sred[tid] = s;
    __syncthreads();
    for (int w = 128; w > 0; w >>= 1) {
        if (tid < w) sred[tid] += sred[tid + w];
        __syncthreads();
    }
    if (tid < NOUT) xo[tid] = x[oidx[tid]];
    __syncthreads();
    const float S = sred[0];

    const int o = blockIdx.x * 256 + tid;
    float dq = 0.f;
    for (int p = 0; p < kpart; ++p) dq += part[(size_t)p * OUTF + o];

    float y = scales[o] * dq - zeros[o] * S + bias[o];

    const float4* owp = (const float4*)(ow + (size_t)o * NOUT);
#pragma unroll
    for (int k4 = 0; k4 < NOUT / 4; ++k4) {
        float4 w4 = owp[k4];
        y += w4.x * xo[4 * k4 + 0] + w4.y * xo[4 * k4 + 1]
           + w4.z * xo[4 * k4 + 2] + w4.w * xo[4 * k4 + 3];
    }
    out[o] = y;
}

// ---------------------------------------------------------------------------
extern "C" void kernel_launch(void* const* d_in, const int* in_sizes, int n_in,
                              void* d_out, int out_size, void* d_ws, size_t ws_size,
                              hipStream_t stream) {
    const float*    x      = (const float*)d_in[0];
    const uint32_t* qw     = (const uint32_t*)d_in[1];
    const float*    scales = (const float*)d_in[2];
    const float*    zeros  = (const float*)d_in[3];
    const float*    bias   = (const float*)d_in[4];
    const float*    ow     = (const float*)d_in[5];
    const int*      oidx   = (const int*)d_in[6];
    float*          out    = (float*)d_out;

    // K-split for occupancy: 64 column-blocks x kpart slices. Partials go to
    // workspace; deterministic (no float atomics). Shrink kpart if ws small;
    // last resort kpart=1 writes partials straight into d_out.
    int kpart = 16;
    while (kpart > 1 && ws_size < (size_t)kpart * OUTF * sizeof(float))
        kpart >>= 1;
    float* part = (float*)d_ws;
    if (ws_size < (size_t)OUTF * sizeof(float)) { part = out; kpart = 1; }
    const int kper = INF / kpart;

    dim3 g1(OUTF / COLS_PER_BLOCK, kpart);
    quant3_partial_kernel<<<g1, 256, 0, stream>>>(qw, x, part, kper);
    quant3_finalize_kernel<<<OUTF / 256, 256, 0, stream>>>(
        x, scales, zeros, bias, ow, oidx, part, kpart, out);
}
